// MultiheadSelfAttention_78005196030329
// MI455X (gfx1250) — matrix operand order repaired and checked by reading, not compiled, not run on hardware
//
#include <hip/hip_runtime.h>

// ---------------------------------------------------------------------------
// Multi-head self-attention for MI455X (gfx1250, wave32, WMMA + async-to-LDS).
//   B=4, C=1, L=2048, D_MODEL=D_K=D_V=1024, H=16, head_dim=64.
// All matmuls on v_wmma_f32_16x16x32_f16 (f16 in, f32 accumulate).
// K/V tiles staged to LDS with global_load_async_to_lds_b128 (ASYNCcnt),
// double-buffered so the copy of tile j+1 overlaps compute on tile j.
// Workspace (f16): Qh[64][2048][64] | Kh[64][2048][64] | Vt[64][64][2048] |
//                  O[8192][1024]  -> 4 x 16MB = 64MB.
// ---------------------------------------------------------------------------

typedef __attribute__((ext_vector_type(16))) _Float16 v16h;
typedef __attribute__((ext_vector_type(8)))  _Float16 v8h;
typedef __attribute__((ext_vector_type(8)))  float    v8f;

#define LOG2E 1.44269504088896340736f
#define L_SEQ 2048
#define HEADS 16
#define HD    64
#define LDT   40   // LDS row stride in halves (32 + 8 pad)

static __device__ inline v16h make_frag(v8h lo, v8h hi) {
  return __builtin_shufflevector(lo, hi, 0,1,2,3,4,5,6,7,8,9,10,11,12,13,14,15);
}

// A-fragment (16x32 f16) from row-major memory, row stride `ld` halves.
// ISA 7.12.2: lane l -> row l%16, K = (l/16)*8 + {0..7} and {16..23}.
static __device__ inline v16h load_fragA(const _Float16* base, int ld, int lane) {
  const int row  = lane & 15;
  const int koff = (lane >> 4) << 3;
  const _Float16* p = base + row * ld + koff;
  v8h lo = *(const v8h*)(p);
  v8h hi = *(const v8h*)(p + 16);
  return make_frag(lo, hi);
}

// B-fragment (32x16 f16) given B^T row-major (16 N-rows x 32 K).
// lane l -> N=l%16, K = (l/16)*8 + {0..7} and {16..23} (matches fragA).
static __device__ inline v16h load_fragB(const _Float16* baseT, int ld, int lane) {
  const int row   = lane & 15;
  const int kbase = (lane >> 4) << 3;
  const _Float16* p = baseT + row * ld + kbase;
  v8h lo = *(const v8h*)(p);
  v8h hi = *(const v8h*)(p + 16);
  return make_frag(lo, hi);
}

static __device__ inline v8f wmma(v16h a, v16h b, v8f c) {
  return __builtin_amdgcn_wmma_f32_16x16x32_f16(false, a, false, b, (short)0, c, false, false);
}

// Async global->LDS 16B copy per lane (ASYNCcnt tracked). GV addressing mode.
// LDS byte address = low 32 bits of the flat shared pointer (aperture form).
static __device__ inline void async_b128(const void* lds, const void* g) {
  asm volatile("global_load_async_to_lds_b128 %0, %1, off"
               :: "v"((uint32_t)(uintptr_t)lds), "v"((uint64_t)(uintptr_t)g)
               : "memory");
}
static __device__ inline void wait_async0() {
  asm volatile("s_wait_asynccnt 0" ::: "memory");
}

// ---------------------------------------------------------------------------
// QKV projection: dst = (X[8192x1024] * W^T[1024x1024] + bias) * scale -> f16
// transpose==0 : dst[(bh)*2048 + l][64]   (Q/K, per-head row-major)
// transpose==1 : dst[(bh)*64 + d][2048]   (V, per-head transposed)
// 128x128 tile / 256 threads, BK=32; 8 waves in 4(M)x2(N), 8 WMMAs each/step.
// ---------------------------------------------------------------------------
__global__ __launch_bounds__(256)
void proj_kernel(const float* __restrict__ X, const float* __restrict__ W,
                 const float* __restrict__ bias, _Float16* __restrict__ dst,
                 int transpose, float scale)
{
  __shared__ __align__(16) _Float16 As[128 * LDT];
  __shared__ __align__(16) _Float16 Bs[128 * LDT];

  const int tid  = threadIdx.x;
  const int lane = tid & 31;
  const int w    = tid >> 5;
  const int mw   = w & 3;          // rows mw*32
  const int nw   = w >> 2;         // cols nw*64
  const int row0 = blockIdx.x * 128;
  const int col0 = blockIdx.y * 128;

  const int ldr = tid >> 1;        // staging row 0..127
  const int ldc = (tid & 1) * 16;  // staging col {0,16}

  v8f z = {};
  v8f acc[2][4];
  #pragma unroll
  for (int i = 0; i < 2; ++i)
    #pragma unroll
    for (int jx = 0; jx < 4; ++jx) acc[i][jx] = z;

  for (int k0 = 0; k0 < 1024; k0 += 32) {
    { // stage X tile (f32 -> f16), 16 values per thread
      const float* src = X + (size_t)(row0 + ldr) * 1024 + k0 + ldc;
      _Float16* d = As + ldr * LDT + ldc;
      #pragma unroll
      for (int i = 0; i < 4; ++i) {
        float4 f = *(const float4*)(src + i * 4);
        d[i*4+0]=(_Float16)f.x; d[i*4+1]=(_Float16)f.y;
        d[i*4+2]=(_Float16)f.z; d[i*4+3]=(_Float16)f.w;
      }
    }
    { // stage W tile (B^T rows = output features), f32 -> f16
      const float* src = W + (size_t)(col0 + ldr) * 1024 + k0 + ldc;
      _Float16* d = Bs + ldr * LDT + ldc;
      #pragma unroll
      for (int i = 0; i < 4; ++i) {
        float4 f = *(const float4*)(src + i * 4);
        d[i*4+0]=(_Float16)f.x; d[i*4+1]=(_Float16)f.y;
        d[i*4+2]=(_Float16)f.z; d[i*4+3]=(_Float16)f.w;
      }
    }
    __syncthreads();

    v16h a0 = load_fragA(As + (mw * 32)      * LDT, LDT, lane);
    v16h a1 = load_fragA(As + (mw * 32 + 16) * LDT, LDT, lane);
    v16h b0 = load_fragB(Bs + (nw * 64)      * LDT, LDT, lane);
    v16h b1 = load_fragB(Bs + (nw * 64 + 16) * LDT, LDT, lane);
    v16h b2 = load_fragB(Bs + (nw * 64 + 32) * LDT, LDT, lane);
    v16h b3 = load_fragB(Bs + (nw * 64 + 48) * LDT, LDT, lane);
    acc[0][0] = wmma(a0, b0, acc[0][0]);
    acc[0][1] = wmma(a0, b1, acc[0][1]);
    acc[0][2] = wmma(a0, b2, acc[0][2]);
    acc[0][3] = wmma(a0, b3, acc[0][3]);
    acc[1][0] = wmma(a1, b0, acc[1][0]);
    acc[1][1] = wmma(a1, b1, acc[1][1]);
    acc[1][2] = wmma(a1, b2, acc[1][2]);
    acc[1][3] = wmma(a1, b3, acc[1][3]);

    __syncthreads();
  }

  // Epilogue: C layout row M = r + 8*(lane/16), col N = lane%16.
  #pragma unroll
  for (int am = 0; am < 2; ++am) {
    const int rbase = row0 + mw * 32 + am * 16 + ((lane >> 4) << 3);
    #pragma unroll
    for (int bn = 0; bn < 4; ++bn) {
      const int col = col0 + nw * 64 + bn * 16 + (lane & 15);
      const float bv = bias[col];
      const int h = col >> 6, d = col & 63;
      #pragma unroll
      for (int r = 0; r < 8; ++r) {
        const int row = rbase + r;
        const int b = row >> 11, l = row & 2047;
        const float val = (acc[am][bn][r] + bv) * scale;
        size_t idx;
        if (transpose) idx = ((size_t)(b * HEADS + h) * HD + d) * L_SEQ + l;
        else           idx = ((size_t)(b * HEADS + h) * L_SEQ + l) * HD + d;
        dst[idx] = (_Float16)val;
      }
    }
  }
}

// ---------------------------------------------------------------------------
// Flash attention: grid (L/128, B*H). 8 waves x 16 query rows each.
// K/V 32-key tiles staged once per block into LDS via async copies,
// double-buffered; online softmax in exp2 domain (log2e folded into Q).
// ---------------------------------------------------------------------------
__global__ __launch_bounds__(256)
void attn_kernel(const _Float16* __restrict__ Qh, const _Float16* __restrict__ Kh,
                 const _Float16* __restrict__ Vt, _Float16* __restrict__ O)
{
  __shared__ __align__(16) _Float16 Ktile[2][32 * 64];  // [key][dk]
  __shared__ __align__(16) _Float16 Vtile[2][64 * 32];  // [dv][key]
  __shared__ __align__(16) _Float16 Plds[8][16 * 32];   // per-wave P transpose

  const int tid  = threadIdx.x;
  const int lane = tid & 31;
  const int w    = tid >> 5;
  const int bh   = blockIdx.y;
  const int qrow0 = blockIdx.x * 128 + w * 16;

  const _Float16* Qb = Qh + (size_t)bh * L_SEQ * HD;
  const _Float16* Kb = Kh + (size_t)bh * L_SEQ * HD;
  const _Float16* Vb = Vt + (size_t)bh * HD * L_SEQ;
  _Float16* P = &Plds[w][0];

  // Per-thread staging slots: K: row=tid/8 (0..31), chunk=tid%8 of 8 halves.
  //                           V: row=tid/4 (0..63), chunk=tid%4 of 8 halves.
  const int krow = tid >> 3, kch = (tid & 7) * 8;
  const int vrow = tid >> 2, vch = (tid & 3) * 8;

  // Prologue: stage key-block 0 into buffer 0.
  async_b128(&Ktile[0][krow * 64 + kch], Kb + (size_t)krow * HD + kch);
  async_b128(&Vtile[0][vrow * 32 + vch], Vb + (size_t)vrow * L_SEQ + vch);

  // Persistent Q fragments: dk 0..31 and 32..63.
  v16h qa0 = load_fragA(Qb + (size_t)qrow0 * HD,      HD, lane);
  v16h qa1 = load_fragA(Qb + (size_t)qrow0 * HD + 32, HD, lane);

  float m[8], lsum[8];
  v8f o0 = {}, o1 = {}, o2 = {}, o3 = {};
  #pragma unroll
  for (int r = 0; r < 8; ++r) { m[r] = -1e30f; lsum[r] = 0.f; }

  const int half = lane >> 4;
  const int coln = lane & 15;

  for (int jb = 0; jb < L_SEQ / 32; ++jb) {
    const int cur = jb & 1;
    wait_async0();       // buffer `cur` landed (per-wave ASYNCcnt)
    __syncthreads();     // all waves see it; readers of `nxt` (=jb-1's cur) done

    if (jb + 1 < L_SEQ / 32) {  // overlap next tile copy with compute
      const int nxt = cur ^ 1;
      const size_t j2 = (size_t)(jb + 1) * 32;
      async_b128(&Ktile[nxt][krow * 64 + kch], Kb + (j2 + krow) * HD + kch);
      async_b128(&Vtile[nxt][vrow * 32 + vch], Vb + (size_t)vrow * L_SEQ + j2 + vch);
    }

    const _Float16* Kt = &Ktile[cur][0];
    const _Float16* Vw = &Vtile[cur][0];

    // S = Q * K^T for keys jb*32..+31 (already in log2e units via Q scale).
    v16h kb00 = load_fragB(Kt,                64, lane);  // keys 0-15, dk 0-31
    v16h kb10 = load_fragB(Kt + 32,           64, lane);  // keys 0-15, dk 32-63
    v16h kb01 = load_fragB(Kt + 16 * 64,      64, lane);  // keys 16-31
    v16h kb11 = load_fragB(Kt + 16 * 64 + 32, 64, lane);

    v8f s0 = {}, s1 = {};
    s0 = wmma(qa0, kb00, s0);
    s0 = wmma(qa1, kb10, s0);
    s1 = wmma(qa0, kb01, s1);
    s1 = wmma(qa1, kb11, s1);

    // Online softmax; row M = r + 8*half lives in one 16-lane half.
    #pragma unroll
    for (int r = 0; r < 8; ++r) {
      float a = s0[r], b = s1[r];
      float mx = fmaxf(a, b);
      mx = fmaxf(mx, __shfl_xor(mx, 1));
      mx = fmaxf(mx, __shfl_xor(mx, 2));
      mx = fmaxf(mx, __shfl_xor(mx, 4));
      mx = fmaxf(mx, __shfl_xor(mx, 8));
      const float mnew = fmaxf(m[r], mx);
      const float corr = exp2f(m[r] - mnew);
      m[r] = mnew;
      const float p0 = exp2f(a - mnew);
      const float p1 = exp2f(b - mnew);
      s0[r] = p0; s1[r] = p1;
      float rs = p0 + p1;
      rs += __shfl_xor(rs, 1);
      rs += __shfl_xor(rs, 2);
      rs += __shfl_xor(rs, 4);
      rs += __shfl_xor(rs, 8);
      lsum[r] = lsum[r] * corr + rs;
      o0[r] *= corr; o1[r] *= corr; o2[r] *= corr; o3[r] *= corr;
    }

    // P: C-layout -> row-major 16x32 in per-wave LDS, reload as A fragment.
    #pragma unroll
    for (int r = 0; r < 8; ++r) {
      const int row = r + 8 * half;
      P[row * 32 + coln]      = (_Float16)s0[r];
      P[row * 32 + 16 + coln] = (_Float16)s1[r];
    }
    asm volatile("s_wait_dscnt 0" ::: "memory");
    v16h pa = load_fragA(P, 32, lane);

    // O += P * V : B^T rows are V^T dv-rows in LDS (stride 32 keys).
    v16h vb0 = load_fragB(Vw,           32, lane);
    v16h vb1 = load_fragB(Vw + 16 * 32, 32, lane);
    v16h vb2 = load_fragB(Vw + 32 * 32, 32, lane);
    v16h vb3 = load_fragB(Vw + 48 * 32, 32, lane);
    o0 = wmma(pa, vb0, o0);
    o1 = wmma(pa, vb1, o1);
    o2 = wmma(pa, vb2, o2);
    o3 = wmma(pa, vb3, o3);
  }

  // Normalize and store to O[B][L][1024] (head-concat layout) as f16.
  const int b = bh >> 4, h = bh & 15;
  _Float16* outp = O + ((size_t)b * L_SEQ + qrow0) * 1024 + h * HD;
  float inv[8];
  #pragma unroll
  for (int r = 0; r < 8; ++r) inv[r] = 1.0f / lsum[r];
  #pragma unroll
  for (int d = 0; d < 4; ++d) {
    v8f oo = (d == 0) ? o0 : (d == 1) ? o1 : (d == 2) ? o2 : o3;
    #pragma unroll
    for (int r = 0; r < 8; ++r) {
      const int row = r + 8 * half;
      outp[(size_t)row * 1024 + d * 16 + coln] = (_Float16)(oo[r] * inv[r]);
    }
  }
}

// ---------------------------------------------------------------------------
// Output projection: out(f32) = O(f16)[8192x1024] * We^T + be
// A tile staged with async-to-LDS (overlaps f32->f16 weight conversion).
// ---------------------------------------------------------------------------
__global__ __launch_bounds__(256)
void outproj_kernel(const _Float16* __restrict__ A, const float* __restrict__ W,
                    const float* __restrict__ bias, float* __restrict__ out)
{
  __shared__ __align__(16) _Float16 As[128 * LDT];
  __shared__ __align__(16) _Float16 Bs[128 * LDT];

  const int tid  = threadIdx.x;
  const int lane = tid & 31;
  const int w    = tid >> 5;
  const int mw   = w & 3;
  const int nw   = w >> 2;
  const int row0 = blockIdx.x * 128;
  const int col0 = blockIdx.y * 128;

  const int ldr = tid >> 1;
  const int ldc = (tid & 1) * 16;

  v8f z = {};
  v8f acc[2][4];
  #pragma unroll
  for (int i = 0; i < 2; ++i)
    #pragma unroll
    for (int jx = 0; jx < 4; ++jx) acc[i][jx] = z;

  for (int k0 = 0; k0 < 1024; k0 += 32) {
    { // A tile already f16: async copy 2x16B per thread into LDS
      const _Float16* src = A + (size_t)(row0 + ldr) * 1024 + k0 + ldc;
      _Float16* d = As + ldr * LDT + ldc;
      async_b128(d,     src);
      async_b128(d + 8, src + 8);
    }
    { // We tile with f32 -> f16 conversion (VALU, overlaps the async copy)
      const float* src = W + (size_t)(col0 + ldr) * 1024 + k0 + ldc;
      _Float16* d = Bs + ldr * LDT + ldc;
      #pragma unroll
      for (int i = 0; i < 4; ++i) {
        float4 f = *(const float4*)(src + i * 4);
        d[i*4+0]=(_Float16)f.x; d[i*4+1]=(_Float16)f.y;
        d[i*4+2]=(_Float16)f.z; d[i*4+3]=(_Float16)f.w;
      }
    }
    wait_async0();
    __syncthreads();

    v16h a0 = load_fragA(As + (mw * 32)      * LDT, LDT, lane);
    v16h a1 = load_fragA(As + (mw * 32 + 16) * LDT, LDT, lane);
    v16h b0 = load_fragB(Bs + (nw * 64)      * LDT, LDT, lane);
    v16h b1 = load_fragB(Bs + (nw * 64 + 16) * LDT, LDT, lane);
    v16h b2 = load_fragB(Bs + (nw * 64 + 32) * LDT, LDT, lane);
    v16h b3 = load_fragB(Bs + (nw * 64 + 48) * LDT, LDT, lane);
    acc[0][0] = wmma(a0, b0, acc[0][0]);
    acc[0][1] = wmma(a0, b1, acc[0][1]);
    acc[0][2] = wmma(a0, b2, acc[0][2]);
    acc[0][3] = wmma(a0, b3, acc[0][3]);
    acc[1][0] = wmma(a1, b0, acc[1][0]);
    acc[1][1] = wmma(a1, b1, acc[1][1]);
    acc[1][2] = wmma(a1, b2, acc[1][2]);
    acc[1][3] = wmma(a1, b3, acc[1][3]);

    __syncthreads();
  }

  #pragma unroll
  for (int am = 0; am < 2; ++am) {
    const int rbase = row0 + mw * 32 + am * 16 + ((lane >> 4) << 3);
    #pragma unroll
    for (int bn = 0; bn < 4; ++bn) {
      const int col = col0 + nw * 64 + bn * 16 + (lane & 15);
      const float bv = bias[col];
      #pragma unroll
      for (int r = 0; r < 8; ++r) {
        out[(size_t)(rbase + r) * 1024 + col] = acc[am][bn][r] + bv;
      }
    }
  }
}

// ---------------------------------------------------------------------------
extern "C" void kernel_launch(void* const* d_in, const int* in_sizes, int n_in,
                              void* d_out, int out_size, void* d_ws, size_t ws_size,
                              hipStream_t stream) {
  const float* query = (const float*)d_in[0];
  const float* key_  = (const float*)d_in[1];
  const float* value = (const float*)d_in[2];
  const float* Wq    = (const float*)d_in[3];
  const float* bq    = (const float*)d_in[4];
  const float* Wk    = (const float*)d_in[5];
  const float* bk    = (const float*)d_in[6];
  const float* Wv    = (const float*)d_in[7];
  const float* bv    = (const float*)d_in[8];
  const float* We    = (const float*)d_in[9];
  const float* be    = (const float*)d_in[10];
  float* out = (float*)d_out;

  char* ws = (char*)d_ws;
  const size_t SEG = (size_t)16 * 1024 * 1024;   // 16 MB per f16 tensor
  _Float16* Qh = (_Float16*)(ws + 0 * SEG);
  _Float16* Kh = (_Float16*)(ws + 1 * SEG);
  _Float16* Vt = (_Float16*)(ws + 2 * SEG);
  _Float16* Oc = (_Float16*)(ws + 3 * SEG);      // total 64 MB of d_ws

  dim3 gemm_grid(64, 8), block(256);
  // scores scaled by 1/sqrt(D_K)=1/32; also fold log2(e) for exp2-domain softmax.
  proj_kernel<<<gemm_grid, block, 0, stream>>>(query, Wq, bq, Qh, 0, 0.03125f * LOG2E);
  proj_kernel<<<gemm_grid, block, 0, stream>>>(key_,  Wk, bk, Kh, 0, 1.0f);
  proj_kernel<<<gemm_grid, block, 0, stream>>>(value, Wv, bv, Vt, 1, 1.0f);
  attn_kernel<<<dim3(16, 64), block, 0, stream>>>(Qh, Kh, Vt, Oc);
  outproj_kernel<<<gemm_grid, block, 0, stream>>>(Oc, We, be, out);
}
